// SoAGREE_22342419874471
// MI455X (gfx1250) — matrix operand order repaired and checked by reading, not compiled, not run on hardware
//
#include <hip/hip_runtime.h>
#include <stdint.h>

typedef __bf16 v16bf __attribute__((ext_vector_type(16)));
typedef float  v8f   __attribute__((ext_vector_type(8)));
typedef int    v4i   __attribute__((ext_vector_type(4)));

#define M_DIM 16
#define F_DIM 32
#define D_DIM 64

// ---- CDNA5 async global->LDS copy (guarded; falls back to ld+st) ----------
#if defined(__HIP_DEVICE_COMPILE__) && !defined(DISABLE_ASYNC)
  #if __has_builtin(__builtin_amdgcn_global_load_async_to_lds_b128)
    #define HAVE_ASYNC 1
  #endif
#endif

#if defined(HAVE_ASYNC)
typedef __attribute__((address_space(1))) v4i gv4i_t;   // global int4
typedef __attribute__((address_space(3))) v4i lv4i_t;   // LDS int4
__device__ __forceinline__ void async_copy16(const void* g, void* l) {
  __builtin_amdgcn_global_load_async_to_lds_b128(
      (gv4i_t*)(unsigned long long)g, (lv4i_t*)l, 0, 0);
}
__device__ __forceinline__ void wait_async() {
#if __has_builtin(__builtin_amdgcn_s_wait_asynccnt)
  __builtin_amdgcn_s_wait_asynccnt(0);
#else
  asm volatile("s_wait_asynccnt 0" ::: "memory");
#endif
}
#else
__device__ __forceinline__ void async_copy16(const void* g, void* l) {
  *(float4*)l = *(const float4*)g;
}
__device__ __forceinline__ void wait_async() {}
#endif

// ---- Pack FULL W1 (128 rows) of both attention MLPs into bf16 B-frags -----
// Layout per table: [tile t(4)][lane(32)][elem(16)];  lane l: N = l&15,
// element e: K = 32*t + 16*(l>>4) + e  (contiguous k order, same as A via a_kk).
__global__ __launch_bounds__(64)
void soagree_prep_weights(const float* __restrict__ fatt_w1,
                          const float* __restrict__ att_w1,
                          __bf16* __restrict__ wsb)
{
  const int tid = threadIdx.x;
  for (int i = 0; i < 64; ++i) {
    int idx   = tid * 64 + i;      // 0..4095
    int table = idx >> 11;         // 0: fatt, 1: att
    int r     = idx & 2047;
    int t     = r >> 9;            // K tile 0..3
    int l     = (r >> 4) & 31;
    int e     = r & 15;
    int n     = l & 15;
    int K     = 32 * t + 16 * (l >> 4) + e;     // 0..127
    const float* W = table ? att_w1 : fatt_w1;
    wsb[idx] = (__bf16)W[K * 16 + n];
  }
}

// A-fragment K index within a 32-wide tile (contiguous: K = 16*half + e, matches B packing)
__device__ __forceinline__ int a_kk(int half, int e) {
  return 16 * half + e;
}

// ---- Phase 1: follow-level attention, one block per (b,m) -----------------
__global__ __launch_bounds__(64)
void soagree_follow_attn(const int* __restrict__ member_ids,
                         const int* __restrict__ follow_ids,
                         const float* __restrict__ user_table,
                         const float* __restrict__ follow_table,
                         const float* __restrict__ fatt_b1,
                         const float* __restrict__ fatt_w2,
                         const float* __restrict__ fatt_b2,
                         const __bf16* __restrict__ wfrag,   // [4][32][16]
                         float* __restrict__ members)
{
  __shared__ float sF[F_DIM * D_DIM]; // 8 KB follow embeddings (f32)
  __shared__ float sU[D_DIM];
  __shared__ float sP[16 * F_DIM];    // transposed products [n][row]
  __shared__ float sS[F_DIM];         // row scores
  __shared__ float sW[F_DIM];         // exp weights

  const int bm   = blockIdx.x;
  const int tid  = threadIdx.x;
  const int lane = tid & 31;
  const int wv   = tid >> 5;

  sU[tid] = user_table[(size_t)member_ids[bm] * D_DIM + tid];

  // gather 32 follow rows (256B each) into LDS; 8 x 16B async per thread
  const int fbase = bm * F_DIM;
  #pragma unroll
  for (int j = 0; j < 8; ++j) {
    int q   = tid + 64 * j;          // float4 slot 0..511
    int row = q >> 4;
    int c4  = q & 15;
    int fid = follow_ids[fbase + row];
    async_copy16(follow_table + (size_t)fid * D_DIM + c4 * 4,
                 &sF[row * D_DIM + c4 * 4]);
  }

  wait_async();
  __syncthreads();  // sF + sU visible

  // ---- score MLP layer 1 on the matrix pipe: K=128 via 4 bf16 WMMAs ----
  // h[r][n] = b1[n] + f_emb[r] @ W1[:64] + u_emb @ W1[64:]
  const int half = lane >> 4;
  const int lr   = lane & 15;
  const int row0 = wv * 16;

  v16bf a0, a1, a2, a3;
  #pragma unroll
  for (int e = 0; e < 16; ++e) {
    int kk = a_kk(half, e);
    a0[e] = (__bf16)sF[(row0 + lr) * D_DIM + kk];        // f_emb K 0..31
    a1[e] = (__bf16)sF[(row0 + lr) * D_DIM + 32 + kk];   // f_emb K 32..63
    a2[e] = (__bf16)sU[kk];                              // u_emb K 0..31 (bcast)
    a3[e] = (__bf16)sU[32 + kk];                         // u_emb K 32..63
  }
  v16bf b0 = *(const v16bf*)(wfrag + 0 * 512 + lane * 16);
  v16bf b1 = *(const v16bf*)(wfrag + 1 * 512 + lane * 16);
  v16bf b2 = *(const v16bf*)(wfrag + 2 * 512 + lane * 16);
  v16bf b3 = *(const v16bf*)(wfrag + 3 * 512 + lane * 16);

  const float b1n = fatt_b1[lr];
  v8f acc;
  #pragma unroll
  for (int v = 0; v < 8; ++v) acc[v] = b1n;    // bias folded into C
  acc = __builtin_amdgcn_wmma_f32_16x16x32_bf16(false, a0, false, b0, (short)0, acc, false, false);
  acc = __builtin_amdgcn_wmma_f32_16x16x32_bf16(false, a1, false, b1, (short)0, acc, false, false);
  acc = __builtin_amdgcn_wmma_f32_16x16x32_bf16(false, a2, false, b2, (short)0, acc, false, false);
  acc = __builtin_amdgcn_wmma_f32_16x16x32_bf16(false, a3, false, b3, (short)0, acc, false, false);

  // products p = relu(h)*w2[n]; transpose via LDS: sP[n][row], 2x b128 store
  const float w2n = fatt_w2[lr];
  {
    float4 p0, p1;
    p0.x = fmaxf(acc[0], 0.f) * w2n;  p0.y = fmaxf(acc[1], 0.f) * w2n;
    p0.z = fmaxf(acc[2], 0.f) * w2n;  p0.w = fmaxf(acc[3], 0.f) * w2n;
    p1.x = fmaxf(acc[4], 0.f) * w2n;  p1.y = fmaxf(acc[5], 0.f) * w2n;
    p1.z = fmaxf(acc[6], 0.f) * w2n;  p1.w = fmaxf(acc[7], 0.f) * w2n;
    float* dst = &sP[lr * F_DIM + row0 + 8 * half];  // 8 consecutive rows
    *(float4*)dst       = p0;
    *(float4*)(dst + 4) = p1;
  }
  __syncthreads();

  // row sums -> scores; exp once per row (32 threads only)
  if (tid < F_DIM) {
    float s = fatt_b2[0];
    #pragma unroll
    for (int n = 0; n < 16; ++n) s += sP[n * F_DIM + tid];
    sS[tid] = s;
  }
  if (tid < F_DIM) {
    float mx = -3.4e38f;
    #pragma unroll
    for (int f = 0; f < F_DIM; ++f) mx = fmaxf(mx, sS[f]);
    sW[tid] = __expf(sS[tid] - mx);
  }
  __syncthreads();

  // softmax-normalized aggregation; thread owns dim d = tid (f32)
  float den = 0.f;
  #pragma unroll
  for (int f = 0; f < F_DIM; ++f) den += sW[f];
  const float inv = 1.f / den;

  float m = sU[tid];
  #pragma unroll
  for (int f = 0; f < F_DIM; ++f) m += sW[f] * inv * sF[f * D_DIM + tid];
  members[(size_t)bm * D_DIM + tid] = m;
}

// ---- Phase 2: member attention + predict head, one block per b ------------
__global__ __launch_bounds__(64)
void soagree_group_head(const int* __restrict__ group_inputs,
                        const int* __restrict__ item_inputs,
                        const float* __restrict__ item_table,
                        const float* __restrict__ group_table,
                        const float* __restrict__ att_b1,
                        const float* __restrict__ att_w2,
                        const float* __restrict__ att_b2,
                        const float* __restrict__ pred_w1,
                        const float* __restrict__ pred_b1,
                        const float* __restrict__ pred_w2,
                        const float* __restrict__ pred_b2,
                        const __bf16* __restrict__ wfrag,   // att B-frags [4][32][16]
                        const float* __restrict__ members,
                        float* __restrict__ out)
{
  __shared__ float sM[M_DIM * D_DIM]; // 4 KB members
  __shared__ float sI[D_DIM];
  __shared__ float sG[D_DIM];
  __shared__ float sP[16 * M_DIM];    // transposed products [n][row]
  __shared__ float sS[M_DIM];
  __shared__ float sW[M_DIM];
  __shared__ float sE[D_DIM];
  __shared__ float sH[8];

  const int b    = blockIdx.x;
  const int tid  = threadIdx.x;
  const int lane = tid & 31;
  const int wv   = tid >> 5;

  sI[tid] = item_table[(size_t)item_inputs[b] * D_DIM + tid];
  sG[tid] = group_table[(size_t)group_inputs[b] * D_DIM + tid];

  const float* mb = members + (size_t)b * M_DIM * D_DIM;
  #pragma unroll
  for (int j = 0; j < 4; ++j) {
    int q = tid + 64 * j;               // float4 slot 0..255
    async_copy16(mb + q * 4, &sM[q * 4]);
  }

  wait_async();
  __syncthreads();  // sM + sI + sG visible

  if (wv == 0) {    // one full wave (EXEC all-ones) runs the K=128 score GEMM
    const int half = lane >> 4;
    const int lr   = lane & 15;
    v16bf a0, a1, a2, a3;
    #pragma unroll
    for (int e = 0; e < 16; ++e) {
      int kk = a_kk(half, e);
      a0[e] = (__bf16)sM[lr * D_DIM + kk];
      a1[e] = (__bf16)sM[lr * D_DIM + 32 + kk];
      a2[e] = (__bf16)sI[kk];
      a3[e] = (__bf16)sI[32 + kk];
    }
    v16bf b0 = *(const v16bf*)(wfrag + 0 * 512 + lane * 16);
    v16bf b1 = *(const v16bf*)(wfrag + 1 * 512 + lane * 16);
    v16bf b2 = *(const v16bf*)(wfrag + 2 * 512 + lane * 16);
    v16bf b3 = *(const v16bf*)(wfrag + 3 * 512 + lane * 16);

    const float b1n = att_b1[lr];
    v8f acc;
    #pragma unroll
    for (int v = 0; v < 8; ++v) acc[v] = b1n;
    acc = __builtin_amdgcn_wmma_f32_16x16x32_bf16(false, a0, false, b0, (short)0, acc, false, false);
    acc = __builtin_amdgcn_wmma_f32_16x16x32_bf16(false, a1, false, b1, (short)0, acc, false, false);
    acc = __builtin_amdgcn_wmma_f32_16x16x32_bf16(false, a2, false, b2, (short)0, acc, false, false);
    acc = __builtin_amdgcn_wmma_f32_16x16x32_bf16(false, a3, false, b3, (short)0, acc, false, false);

    const float w2n = att_w2[lr];
    float4 p0, p1;
    p0.x = fmaxf(acc[0], 0.f) * w2n;  p0.y = fmaxf(acc[1], 0.f) * w2n;
    p0.z = fmaxf(acc[2], 0.f) * w2n;  p0.w = fmaxf(acc[3], 0.f) * w2n;
    p1.x = fmaxf(acc[4], 0.f) * w2n;  p1.y = fmaxf(acc[5], 0.f) * w2n;
    p1.z = fmaxf(acc[6], 0.f) * w2n;  p1.w = fmaxf(acc[7], 0.f) * w2n;
    float* dst = &sP[lr * M_DIM + 8 * half];   // rows v+8*half contiguous
    *(float4*)dst       = p0;
    *(float4*)(dst + 4) = p1;
  }
  __syncthreads();

  if (tid < M_DIM) {
    float s = att_b2[0];
    #pragma unroll
    for (int n = 0; n < 16; ++n) s += sP[n * M_DIM + tid];
    sS[tid] = s;
  }
  if (tid < M_DIM) {
    float mx = -3.4e38f;
    #pragma unroll
    for (int m = 0; m < M_DIM; ++m) mx = fmaxf(mx, sS[m]);
    sW[tid] = __expf(sS[tid] - mx);
  }
  __syncthreads();

  float den = 0.f;
  #pragma unroll
  for (int m = 0; m < M_DIM; ++m) den += sW[m];
  const float inv = 1.f / den;

  float g = sG[tid];
  #pragma unroll
  for (int m = 0; m < M_DIM; ++m) g += sW[m] * inv * sM[m * D_DIM + tid];
  sE[tid] = g;
  __syncthreads();

  // predict head: new = [g*i, g, i] (192) -> 8 -> relu -> 1 -> sigmoid
  if (tid < 8) {
    float h = pred_b1[tid];
    for (int k = 0; k < D_DIM; ++k) {
      h += (sE[k] * sI[k]) * pred_w1[k * 8 + tid];
      h += sE[k] * pred_w1[(D_DIM + k) * 8 + tid];
      h += sI[k] * pred_w1[(2 * D_DIM + k) * 8 + tid];
    }
    sH[tid] = fmaxf(h, 0.f);
  }
  __syncthreads();
  if (tid == 0) {
    float z = pred_b2[0];
    #pragma unroll
    for (int j = 0; j < 8; ++j) z += sH[j] * pred_w2[j];
    out[b] = 1.f / (1.f + __expf(-z));
  }
}

extern "C" void kernel_launch(void* const* d_in, const int* in_sizes, int n_in,
                              void* d_out, int out_size, void* d_ws, size_t ws_size,
                              hipStream_t stream) {
  const int*   group_inputs = (const int*)d_in[0];
  const int*   item_inputs  = (const int*)d_in[1];
  const int*   member_ids   = (const int*)d_in[2];
  const int*   follow_ids   = (const int*)d_in[3];
  const float* user_table   = (const float*)d_in[4];
  const float* item_table   = (const float*)d_in[5];
  const float* group_table  = (const float*)d_in[6];
  const float* follow_table = (const float*)d_in[7];
  const float* fatt_w1 = (const float*)d_in[8];
  const float* fatt_b1 = (const float*)d_in[9];
  const float* fatt_w2 = (const float*)d_in[10];
  const float* fatt_b2 = (const float*)d_in[11];
  const float* att_w1  = (const float*)d_in[12];
  const float* att_b1  = (const float*)d_in[13];
  const float* att_w2  = (const float*)d_in[14];
  const float* att_b2  = (const float*)d_in[15];
  const float* pred_w1 = (const float*)d_in[16];
  const float* pred_b1 = (const float*)d_in[17];
  const float* pred_w2 = (const float*)d_in[18];
  const float* pred_b2 = (const float*)d_in[19];

  const int B = in_sizes[0];
  (void)n_in; (void)out_size; (void)ws_size;

  __bf16* wsb    = (__bf16*)d_ws;                  // 2 tables x [4][32][16] (8 KB)
  float*  membrs = (float*)((char*)d_ws + 8192);   // [B, M, 64] f32

  soagree_prep_weights<<<1, 64, 0, stream>>>(fatt_w1, att_w1, wsb);
  soagree_follow_attn<<<B * M_DIM, 64, 0, stream>>>(
      member_ids, follow_ids, user_table, follow_table,
      fatt_b1, fatt_w2, fatt_b2, wsb, membrs);
  soagree_group_head<<<B, 64, 0, stream>>>(
      group_inputs, item_inputs, item_table, group_table,
      att_b1, att_w2, att_b2,
      pred_w1, pred_b1, pred_w2, pred_b2,
      wsb + 2048, membrs, (float*)d_out);
}
